// TemporalCrossTransformer_20890720927794
// MI455X (gfx1250) — matrix operand order repaired and checked by reading, not compiled, not run on hardware
//
#include <hip/hip_runtime.h>

// ---------------- problem constants ----------------
#define SEQ   8
#define DM    2048          // d_model
#define DK    4096          // 2*d_model (concat tuple)
#define ODIM  1152          // out_dim
#define WAY   5
#define SHOT  5
#define NS    25            // n_support
#define NQ    200           // n_queries
#define NL    28            // temporal tuples
#define NROWS 6300          // 225*28 (support+query tuple rows)
#define MQ    5600          // 200*28 query rows
#define NSUP  700           // 25*28 support rows
#define KSL   140           // SHOT*NL
#define KPAD  160           // padded to multiple of 32
#define SLD   800           // scores/attn row stride = WAY*KPAD

#define LDSB_STRIDE 40      // bf16 elements per LDS B row (32 used + 8 pad, breaks bank conflicts)

typedef __attribute__((ext_vector_type(16))) __bf16 bfx16;
typedef __attribute__((ext_vector_type(8)))  float  v8f;

union Frag { uint4 u[2]; bfx16 v; };

// async-copy builtin parameter types (from hipcc diagnostic):
//   param0: int vector_size(16), address_space(1)  (global source)
//   param1: int vector_size(16), address_space(3)  (LDS destination)
typedef int v4i __attribute__((vector_size(16)));
typedef __attribute__((address_space(1))) v4i gv4i;
typedef __attribute__((address_space(3))) v4i lv4i;

#if __has_builtin(__builtin_amdgcn_global_load_async_to_lds_b128)
#define HAVE_ASYNC_LDS 1
#else
#define HAVE_ASYNC_LDS 0
#endif

#if __has_builtin(__builtin_amdgcn_s_wait_asynccnt)
#define WAIT_ASYNC(n) __builtin_amdgcn_s_wait_asynccnt(n)
#else
#define WAIT_ASYNC(n) asm volatile("s_wait_asynccnt %0" ::"i"(n) : "memory")
#endif

__constant__ int c_t0[NL] = {0,0,0,0,0,0,0,1,1,1,1,1,1,2,2,2,2,2,3,3,3,3,4,4,4,5,5,6};
__constant__ int c_t1[NL] = {1,2,3,4,5,6,7,2,3,4,5,6,7,3,4,5,6,7,4,5,6,7,5,6,7,6,7,7};
__constant__ int c_pa[15] = {0,0,0,0,0,1,1,1,1,2,2,2,3,3,4};
__constant__ int c_pb[15] = {0,1,2,3,4,1,2,3,4,2,3,4,3,4,4};

__device__ __forceinline__ v8f v8f_splat(float x) {
    v8f r;
#pragma unroll
    for (int i = 0; i < 8; ++i) r[i] = x;
    return r;
}

// ---------------- stage 0: casts / feature build ----------------
__global__ void cast_weights_kernel(const float* __restrict__ kw, const float* __restrict__ vw,
                                    __bf16* __restrict__ kwb, __bf16* __restrict__ vwb, int n) {
    int i = blockIdx.x * blockDim.x + threadIdx.x;
    if (i < n) { kwb[i] = (__bf16)kw[i]; vwb[i] = (__bf16)vw[i]; }
}

// xpe[n][t][d] = (input + PE) in bf16 ; n<25 -> support, else query
__global__ void build_xpe_kernel(const float* __restrict__ sup, const float* __restrict__ qry,
                                 __bf16* __restrict__ xpe) {
    int i = blockIdx.x * blockDim.x + threadIdx.x;
    if (i >= (NS + NQ) * SEQ * DM) return;
    int d = i % DM;
    int t = (i / DM) % SEQ;
    int n = i / (DM * SEQ);
    int dd = d & ~1;
    float freq = __expf((float)dd * (-9.210340371976184f / (float)DM)); // -ln(10000)/d
    float ang  = (float)t * freq;
    float pe   = 0.1f * ((d & 1) ? __cosf(ang) : __sinf(ang));
    float x    = (n < NS) ? sup[i] : qry[i - NS * SEQ * DM];
    xpe[i] = (__bf16)(x + pe);
}

// ---------------- stage 1: projection GEMM (implicit tuple gather) ----------------
// out[row, o] = cat(xpe[n][t0], xpe[n][t1]) . W[o,:] + bias[o]
// 128-thread block = 4 waves. Block tile 64(M) x 64(N); each wave owns 16 M-rows.
// Weight tile (64 N x 32 K) is staged into LDS once per K-step and shared by all
// 4 waves, double-buffered via GLOBAL_LOAD_ASYNC_TO_LDS_B128 (ASYNCcnt) when available.
__global__ __launch_bounds__(128) void proj_gemm_kernel(const __bf16* __restrict__ xpe,
                                                        const __bf16* __restrict__ W,
                                                        const float* __restrict__ bias,
                                                        float* __restrict__ out) {
    __shared__ __bf16 ldsB[2][64 * LDSB_STRIDE];

    const int tid  = threadIdx.x;
    const int wave = tid >> 5, lane = tid & 31;
    const int half = lane >> 4, mr = lane & 15;
    const int nb = blockIdx.x * 64;
    const int rowBase = blockIdx.y * 64 + wave * 16;

    int row = rowBase + mr; if (row >= NROWS) row = NROWS - 1;
    const int n0 = row / NL, l = row % NL;
    const __bf16* a0 = xpe + ((size_t)n0 * SEQ + c_t0[l]) * DM;
    const __bf16* a1 = xpe + ((size_t)n0 * SEQ + c_t1[l]) * DM;

    v8f acc[4];
#pragma unroll
    for (int j = 0; j < 4; ++j) acc[j] = v8f_splat(bias[nb + 16 * j + mr]);

    // per-thread B-stage assignment: 64 rows x 4 16B-chunks = 256 chunks, 2 per thread
    const int bRow = tid >> 1;          // 0..63
    const int bOff = (tid & 1) * 8;     // element offset 0 or 8 (second chunk at +16)
    const __bf16* bSrc = W + (size_t)(nb + bRow) * DK + bOff;
    __bf16* bDstBase = &ldsB[0][0] + bRow * LDSB_STRIDE + bOff;

    auto compute_step = [&](const __bf16* ab, int buf) {
        Frag a;
        a.u[0] = *(const uint4*)(ab + 16 * half);
        a.u[1] = *(const uint4*)(ab + 16 * half + 8);
        const __bf16* lb = &ldsB[buf][0];
#pragma unroll
        for (int j = 0; j < 4; ++j) {
            const __bf16* bp = lb + (16 * j + mr) * LDSB_STRIDE + 16 * half;
            Frag b;
            b.u[0] = ((const uint4*)bp)[0];
            b.u[1] = ((const uint4*)bp)[1];
            acc[j] = __builtin_amdgcn_wmma_f32_16x16x32_bf16(false, a.v, false, b.v,
                                                             (short)0, acc[j], false, false);
        }
    };

#if HAVE_ASYNC_LDS
    {
        // prologue: stage kb=0 into buffer 0
        __builtin_amdgcn_global_load_async_to_lds_b128(
            (gv4i*)(bSrc), (lv4i*)(bDstBase), 0, 0);
        __builtin_amdgcn_global_load_async_to_lds_b128(
            (gv4i*)(bSrc + 16), (lv4i*)(bDstBase + 16), 0, 0);
        for (int i = 0; i < DK / 32; ++i) {
            const int kb = i * 32;
            if (i + 1 < DK / 32) {
                const int kn = kb + 32;
                const __bf16* s = bSrc + kn;
                __bf16* d = bDstBase + ((i + 1) & 1) * (64 * LDSB_STRIDE);
                __builtin_amdgcn_global_load_async_to_lds_b128((gv4i*)s, (lv4i*)d, 0, 0);
                __builtin_amdgcn_global_load_async_to_lds_b128((gv4i*)(s + 16), (lv4i*)(d + 16), 0, 0);
                WAIT_ASYNC(2);              // current stage complete, next still in flight
            } else {
                WAIT_ASYNC(0);
            }
            __syncthreads();                // all waves' stage-i data visible
            const __bf16* ab = (kb < DM) ? (a0 + kb) : (a1 + (kb - DM));
            compute_step(ab, i & 1);
            __syncthreads();                // protect buffer (i&1) before it is re-staged
        }
    }
#else
    {
        for (int i = 0; i < DK / 32; ++i) {
            const int kb = i * 32;
            uint4 r0 = *(const uint4*)(bSrc + kb);
            uint4 r1 = *(const uint4*)(bSrc + kb + 16);
            __syncthreads();                // previous compute finished reading LDS
            *(uint4*)(bDstBase)      = r0;
            *(uint4*)(bDstBase + 16) = r1;
            __syncthreads();
            const __bf16* ab = (kb < DM) ? (a0 + kb) : (a1 + (kb - DM));
            compute_step(ab, 0);
        }
    }
#endif

#pragma unroll
    for (int j = 0; j < 4; ++j)
#pragma unroll
        for (int r = 0; r < 8; ++r) {
            int m = rowBase + r + 8 * half;
            if (m < NROWS) out[(size_t)m * ODIM + nb + 16 * j + mr] = acc[j][r];
        }
}

// ---------------- stage 2: layernorm -> bf16 ----------------
__global__ void ln_to_bf16_kernel(const float* __restrict__ x, const float* __restrict__ g,
                                  const float* __restrict__ b, __bf16* __restrict__ y) {
    __shared__ float red[256];
    const int row = blockIdx.x, tid = threadIdx.x;
    const float* xr = x + (size_t)row * ODIM;
    float s = 0.f, s2 = 0.f;
    for (int i = tid; i < ODIM; i += 256) { float v = xr[i]; s += v; s2 += v * v; }
    red[tid] = s; __syncthreads();
    for (int st = 128; st > 0; st >>= 1) { if (tid < st) red[tid] += red[tid + st]; __syncthreads(); }
    float mu = red[0] / (float)ODIM; __syncthreads();
    red[tid] = s2; __syncthreads();
    for (int st = 128; st > 0; st >>= 1) { if (tid < st) red[tid] += red[tid + st]; __syncthreads(); }
    float var = red[0] / (float)ODIM - mu * mu;
    float rs = rsqrtf(var + 1e-5f);
    __bf16* yr = y + (size_t)row * ODIM;
    for (int i = tid; i < ODIM; i += 256)
        yr[i] = (__bf16)((xr[i] - mu) * rs * g[i] + b[i]);
}

// ---------------- stage 3: attention scores GEMM ----------------
// scores[m, n] = ks_q[m] . ks_s[n] / sqrt(ODIM); stored at [m][w*160 + (n%140)]
__global__ __launch_bounds__(32) void scores_gemm_kernel(const __bf16* __restrict__ ks,
                                                         float* __restrict__ scores) {
    const int lane = threadIdx.x, half = lane >> 4, mr = lane & 15;
    const int nbBase = blockIdx.x * 16;
    const int rowBase = blockIdx.y * 16;
    int nL = nbBase + mr; if (nL >= NSUP) nL = NSUP - 1;
    const __bf16* arow = ks + (size_t)(NSUP + rowBase + mr) * ODIM;  // query rows start at 700
    const __bf16* brow = ks + (size_t)nL * ODIM;                     // support rows
    v8f acc = v8f_splat(0.f);
    for (int kb = 0; kb < ODIM; kb += 32) {
        Frag a, b;
        a.u[0] = *(const uint4*)(arow + kb + 16 * half);
        a.u[1] = *(const uint4*)(arow + kb + 16 * half + 8);
        const __bf16* bp = brow + kb + 16 * half;
        b.u[0] = ((const uint4*)bp)[0];
        b.u[1] = ((const uint4*)bp)[1];
        acc = __builtin_amdgcn_wmma_f32_16x16x32_bf16(false, a.v, false, b.v,
                                                      (short)0, acc, false, false);
    }
    const float scale = 0.029462782549439476f; // 1/sqrt(1152)
    const int n = nbBase + mr;
    if (n < NSUP) {
        const int w = n / KSL, r = n % KSL;
#pragma unroll
        for (int rr = 0; rr < 8; ++rr) {
            int m = rowBase + rr + 8 * half;
            scores[(size_t)m * SLD + w * KPAD + r] = acc[rr] * scale;
        }
    }
}

// ---------------- stage 4: softmax over 140, write bf16 attn (zero-padded to 160) ------
__global__ void softmax_attn_kernel(const float* __restrict__ scores, __bf16* __restrict__ attn) {
    __shared__ float red[256];
    const int m = blockIdx.x, w = blockIdx.y, tid = threadIdx.x;
    const float* src = scores + (size_t)m * SLD + w * KPAD;
    float x = (tid < KSL) ? src[tid] : -3.4e38f;
    red[tid] = x; __syncthreads();
    for (int st = 128; st > 0; st >>= 1) { if (tid < st) red[tid] = fmaxf(red[tid], red[tid + st]); __syncthreads(); }
    float mx = red[0]; __syncthreads();
    float e = (tid < KSL) ? __expf(x - mx) : 0.f;
    red[tid] = e; __syncthreads();
    for (int st = 128; st > 0; st >>= 1) { if (tid < st) red[tid] += red[tid + st]; __syncthreads(); }
    float inv = 1.f / red[0];
    if (tid < KPAD)
        attn[(size_t)m * SLD + w * KPAD + tid] = (__bf16)((tid < KSL) ? e * inv : 0.f);
}

// ---------------- stage 5: transpose class V-features to [w][o][kpad] bf16 -------------
__global__ void build_vsT_kernel(const float* __restrict__ vs, __bf16* __restrict__ vsT) {
    int i = blockIdx.x * blockDim.x + threadIdx.x;
    if (i >= WAY * ODIM * KPAD) return;
    int k = i % KPAD;
    int o = (i / KPAD) % ODIM;
    int w = i / (KPAD * ODIM);
    float v = (k < KSL) ? vs[(size_t)(w * KSL + k) * ODIM + o] : 0.f;
    vsT[i] = (__bf16)v;
}

// ---------------- stage 6: prototype GEMM: proto[w,m,o] = attn[w,m,:] @ vsT[w,:,o] -----
__global__ __launch_bounds__(32) void proto_gemm_kernel(const __bf16* __restrict__ attn,
                                                        const __bf16* __restrict__ vsT,
                                                        float* __restrict__ proto) {
    const int lane = threadIdx.x, half = lane >> 4, mr = lane & 15;
    const int nb = blockIdx.x * 64;
    const int rowBase = blockIdx.y * 16;
    const int w = blockIdx.z;
    const __bf16* arow = attn + (size_t)(rowBase + mr) * SLD + w * KPAD;
    v8f acc[4];
#pragma unroll
    for (int j = 0; j < 4; ++j) acc[j] = v8f_splat(0.f);
    for (int kb = 0; kb < KPAD; kb += 32) {
        Frag a;
        a.u[0] = *(const uint4*)(arow + kb + 16 * half);
        a.u[1] = *(const uint4*)(arow + kb + 16 * half + 8);
#pragma unroll
        for (int j = 0; j < 4; ++j) {
            const __bf16* bp = vsT + ((size_t)w * ODIM + nb + 16 * j + mr) * KPAD + kb + 16 * half;
            Frag b;
            b.u[0] = ((const uint4*)bp)[0];
            b.u[1] = ((const uint4*)bp)[1];
            acc[j] = __builtin_amdgcn_wmma_f32_16x16x32_bf16(false, a.v, false, b.v,
                                                             (short)0, acc[j], false, false);
        }
    }
    float* orow = proto + (size_t)w * MQ * ODIM;
#pragma unroll
    for (int j = 0; j < 4; ++j)
#pragma unroll
        for (int r = 0; r < 8; ++r) {
            int m = rowBase + r + 8 * half;
            orow[(size_t)m * ODIM + nb + 16 * j + mr] = acc[j][r];
        }
}

// ---------------- stage 7: ori logits ----------------
__global__ void ori_kernel(const float* __restrict__ vs, const float* __restrict__ proto,
                           float* __restrict__ out) {
    __shared__ float red[256];
    const int q = blockIdx.x / WAY, w = blockIdx.x % WAY, tid = threadIdx.x;
    const float* vq = vs + (size_t)(NSUP + q * NL) * ODIM;
    const float* pp = proto + ((size_t)w * MQ + q * NL) * ODIM;
    float s = 0.f;
    for (int i = tid; i < NL * ODIM; i += 256) { float d = vq[i] - pp[i]; s += d * d; }
    red[tid] = s; __syncthreads();
    for (int st = 128; st > 0; st >>= 1) { if (tid < st) red[tid] += red[tid + st]; __syncthreads(); }
    if (tid == 0) out[NQ * WAY * WAY + q * WAY + w] = -red[0] / (float)NL;
}

// ---------------- stage 8: pairwise prototype dot products ----------------
__global__ void sim_dot_kernel(const float* __restrict__ proto, float* __restrict__ num) {
    __shared__ float red[256];
    const int p = blockIdx.x, q = blockIdx.y, tid = threadIdx.x;
    const int a = c_pa[p], b = c_pb[p];
    const float* pa = proto + ((size_t)a * MQ + q * NL) * ODIM;
    const float* pb = proto + ((size_t)b * MQ + q * NL) * ODIM;
    float s = 0.f;
    for (int i = tid; i < NL * ODIM; i += 256) s += pa[i] * pb[i];
    red[tid] = s; __syncthreads();
    for (int st = 128; st > 0; st >>= 1) { if (tid < st) red[tid] += red[tid + st]; __syncthreads(); }
    if (tid == 0) {
        num[q * 25 + a * 5 + b] = red[0];
        num[q * 25 + b * 5 + a] = red[0];
    }
}

__global__ void sim_fin_kernel(const float* __restrict__ num, float* __restrict__ out) {
    const int q = blockIdx.x, tid = threadIdx.x;
    if (tid < 25) {
        const int a = tid / 5, b = tid % 5;
        float naa = num[q * 25 + a * 5 + a];
        float nbb = num[q * 25 + b * 5 + b];
        float den = fmaxf(sqrtf(naa) * sqrtf(nbb), 1e-8f);
        out[q * 25 + tid] = num[q * 25 + tid] / den;
    }
}

// ---------------- host launcher ----------------
extern "C" void kernel_launch(void* const* d_in, const int* in_sizes, int n_in,
                              void* d_out, int out_size, void* d_ws, size_t ws_size,
                              hipStream_t stream) {
    (void)in_sizes; (void)n_in; (void)out_size; (void)ws_size;
    const float* support = (const float*)d_in[0];
    /* d_in[1] = support_labels: sorted equal-shot by construction, not needed */
    const float* queries = (const float*)d_in[2];
    const float* k_w  = (const float*)d_in[3];
    const float* k_b  = (const float*)d_in[4];
    const float* v_w  = (const float*)d_in[5];
    const float* v_b  = (const float*)d_in[6];
    const float* ln_g = (const float*)d_in[7];
    const float* ln_b = (const float*)d_in[8];
    float* out = (float*)d_out;

    char* ws = (char*)d_ws;
    size_t off = 0;
    auto nxt = [&](size_t bytes) -> void* {
        void* p = ws + off;
        off += (bytes + 255) & ~(size_t)255;
        return p;
    };
    __bf16* kw_bf  = (__bf16*)nxt((size_t)ODIM * DK * 2);            // 9.4 MB
    __bf16* vw_bf  = (__bf16*)nxt((size_t)ODIM * DK * 2);            // 9.4 MB
    __bf16* xpe_bf = (__bf16*)nxt((size_t)(NS + NQ) * SEQ * DM * 2); // 7.4 MB
    float*  ks_all = (float*) nxt((size_t)NROWS * ODIM * 4);         // 29 MB
    float*  vs_all = (float*) nxt((size_t)NROWS * ODIM * 4);         // 29 MB
    __bf16* ks_bf  = (__bf16*)nxt((size_t)NROWS * ODIM * 2);         // 14.5 MB
    float*  scores = (float*) nxt((size_t)MQ * SLD * 4);             // 17.9 MB
    __bf16* attn   = (__bf16*)nxt((size_t)MQ * SLD * 2);             // 9 MB
    __bf16* vsT    = (__bf16*)nxt((size_t)WAY * ODIM * KPAD * 2);    // 1.8 MB
    float*  proto  = (float*) nxt((size_t)WAY * MQ * ODIM * 4);      // 129 MB
    float*  num    = (float*) nxt((size_t)NQ * 25 * 4);              // 20 KB

    const int nw = ODIM * DK;                    // 4,718,592
    cast_weights_kernel<<<(nw + 255) / 256, 256, 0, stream>>>(k_w, v_w, kw_bf, vw_bf, nw);

    const int nx = (NS + NQ) * SEQ * DM;         // 3,686,400
    build_xpe_kernel<<<(nx + 255) / 256, 256, 0, stream>>>(support, queries, xpe_bf);

    dim3 gProj(ODIM / 64, (NROWS + 63) / 64);    // (18, 99), 128 threads
    proj_gemm_kernel<<<gProj, 128, 0, stream>>>(xpe_bf, kw_bf, k_b, ks_all);
    proj_gemm_kernel<<<gProj, 128, 0, stream>>>(xpe_bf, vw_bf, v_b, vs_all);

    ln_to_bf16_kernel<<<NROWS, 256, 0, stream>>>(ks_all, ln_g, ln_b, ks_bf);

    dim3 gSc((NSUP + 15) / 16, MQ / 16);         // (44, 350)
    scores_gemm_kernel<<<gSc, 32, 0, stream>>>(ks_bf, scores);

    softmax_attn_kernel<<<dim3(MQ, WAY), 256, 0, stream>>>(scores, attn);

    const int nt = WAY * ODIM * KPAD;            // 921,600
    build_vsT_kernel<<<(nt + 255) / 256, 256, 0, stream>>>(vs_all, vsT);

    proto_gemm_kernel<<<dim3(ODIM / 64, MQ / 16, WAY), 32, 0, stream>>>(attn, vsT, proto);

    ori_kernel<<<NQ * WAY, 256, 0, stream>>>(vs_all, proto, out);

    sim_dot_kernel<<<dim3(15, NQ), 256, 0, stream>>>(proto, num);
    sim_fin_kernel<<<NQ, 32, 0, stream>>>(num, out);
}
